// TreeNet_56478819942411
// MI455X (gfx1250) — matrix operand order repaired and checked by reading, not compiled, not run on hardware
//
#include <hip/hip_runtime.h>
#include <hip/hip_bf16.h>
#include <math.h>

// TreeNet on MI455X (gfx1250):
//  - parallel big GEMM for x@Wx (no sequential dep) with v_wmma_f32_16x16x32_bf16
//  - sequential stack-scan kernel (1 workgroup, 32 waves) for child contributions.
//    Wc (bf16, transposed) lives in 256 KB of *dynamic* LDS (WGP has 320 KB),
//    so per-step B-fragments come from ds_load_b128 instead of global memory;
//    only the child-row gathers (L2-resident: whole memory array is 128 MB
//    inside the 192 MB L2) hit the vector-memory path each step.

#define T_DIM 1023
#define B_DIM 128
#define DIN_  256
#define DOUT_ 256
#define BF_   2

typedef __attribute__((ext_vector_type(16))) __bf16 v16bf;
typedef __attribute__((ext_vector_type(8)))  float  v8f;

// ---- fragment loaders (contiguous K order, same for A and B) -----------------------

// A (16x32 bf16): lane (half = lane>>4, r = lane&15) holds row M=r.
// elements 0..15 -> K = kc*32 + half*16 + e        (contiguous 16)
__device__ inline v16bf load_a_frag_f32(const float* row, int kc, int half) {
    v16bf a = {};
    if (row) {
        const float* p0 = row + kc * 32 + half * 16;
        float4 x0 = *(const float4*)(p0);
        float4 x1 = *(const float4*)(p0 + 4);
        float4 y0 = *(const float4*)(p0 + 8);
        float4 y1 = *(const float4*)(p0 + 12);
        a[0]  = (__bf16)x0.x; a[1]  = (__bf16)x0.y; a[2]  = (__bf16)x0.z; a[3]  = (__bf16)x0.w;
        a[4]  = (__bf16)x1.x; a[5]  = (__bf16)x1.y; a[6]  = (__bf16)x1.z; a[7]  = (__bf16)x1.w;
        a[8]  = (__bf16)y0.x; a[9]  = (__bf16)y0.y; a[10] = (__bf16)y0.z; a[11] = (__bf16)y0.w;
        a[12] = (__bf16)y1.x; a[13] = (__bf16)y1.y; a[14] = (__bf16)y1.z; a[15] = (__bf16)y1.w;
    }
    return a;
}

// B (32x16 bf16): lane holds col N=r, element e -> K = kc*32 + half*16 + e.
// wT is N-major (row n holds 256 contiguous K values): one 32-byte
// contiguous load per lane (global: 2x b128; LDS: 2x ds_load_b128).
__device__ inline v16bf load_b_frag_g(const __hip_bfloat16* wT, int n, int kc, int half, int r) {
    const __bf16* p = (const __bf16*)wT + ((size_t)(n * 16 + r)) * DOUT_ + kc * 32 + half * 16;
    return *(const v16bf*)p;
}

__device__ inline v16bf load_b_frag_lds(const __bf16* wT, int n, int kc, int half, int r) {
    const __bf16* p = wT + (n * 16 + r) * DOUT_ + kc * 32 + half * 16;
    return *(const v16bf*)p;
}

// ---- kernel 0: fp32 -> bf16 transposed weight copies -----------------------

__global__ void convert_weights(const float* __restrict__ Wx, const float* __restrict__ Wc,
                                __hip_bfloat16* __restrict__ WxT, __hip_bfloat16* __restrict__ WcT) {
    int i = blockIdx.x * blockDim.x + threadIdx.x;
    if (i < DIN_ * DOUT_) {
        int n = i >> 8, k = i & 255;
        WxT[i] = __float2bfloat16(Wx[k * DOUT_ + n]);               // WxT[n][k] = Wx[k][n]
    }
    if (i < BF_ * DOUT_ * DOUT_) {
        int m = i >> 16, rem = i & 65535;
        int n = rem >> 8, k = rem & 255;
        WcT[i] = __float2bfloat16(Wc[(m * DOUT_ + k) * DOUT_ + n]); // WcT[m][n][k] = Wc[m][k][n]
    }
}

// ---- kernel 1: M[t] = x_t @ Wx + b  (fully parallel, one block per t) ------

__global__ __launch_bounds__(1024) void xwx_kernel(const float* __restrict__ x,
                                                   const __hip_bfloat16* __restrict__ WxT,
                                                   const float* __restrict__ bias,
                                                   float* __restrict__ M) {
    int t    = blockIdx.x;
    int tid  = threadIdx.x;
    int wave = tid >> 5, lane = tid & 31;
    int half = lane >> 4, r = lane & 15;
    int mtile = wave >> 2;          // 8 row tiles of 16 batch rows
    int n0    = (wave & 3) * 4;     // 4 col tiles per wave

    const float* xrow = x + ((size_t)t * B_DIM + mtile * 16 + r) * DIN_;
    v8f acc[4] = {};
    for (int kc = 0; kc < 8; ++kc) {
        v16bf a = load_a_frag_f32(xrow, kc, half);
#pragma unroll
        for (int j = 0; j < 4; ++j) {
            v16bf b = load_b_frag_g(WxT, n0 + j, kc, half, r);
            acc[j] = __builtin_amdgcn_wmma_f32_16x16x32_bf16(
                false, a, false, b, (short)0, acc[j], false, false);
        }
    }
    float* Mt = M + (size_t)t * B_DIM * DOUT_;
#pragma unroll
    for (int j = 0; j < 4; ++j) {
        int col = (n0 + j) * 16 + r;
        float bv = bias[col];
#pragma unroll
        for (int v = 0; v < 8; ++v) {
            int row = mtile * 16 + half * 8 + v;   // C layout: element v -> row half*8+v
            Mt[row * DOUT_ + col] = acc[j][v] + bv;
        }
    }
}

// ---- kernel 2: sequential stack scan (single workgroup, 32 waves) ----------
// Dynamic LDS: WcT bf16 [BF][DOUT][DOUT] = 256 KB (gfx1250 WGP: 320 KB max).

__global__ __launch_bounds__(1024) void scan_kernel(const int* __restrict__ arities,
                                                    const __hip_bfloat16* __restrict__ WcT,
                                                    float* __restrict__ M,
                                                    int* __restrict__ stack,
                                                    float* __restrict__ out) {
    extern __shared__ __align__(32) char smem[];
    __bf16* sWc = (__bf16*)smem;                    // [BF_][DOUT_][DOUT_]

    __shared__ int s_idx[BF_][B_DIM];
    __shared__ int s_valid[BF_][B_DIM];
    __shared__ int s_any[BF_];
    __shared__ int s_root[B_DIM];

    int tid = threadIdx.x;

    // stage WcT (bf16, N-major) global -> LDS, 16 B per thread per iter
    {
        const uint4* src = (const uint4*)WcT;
        uint4*       dst = (uint4*)sWc;
        const int nvec = BF_ * DOUT_ * DOUT_ * 2 / 16;   // 16384
        for (int i = tid; i < nvec; i += 1024) dst[i] = src[i];
    }

    // zero the stack region (ws is poisoned by the harness)
    for (int i = tid; i < B_DIM * (T_DIM + BF_); i += 1024) stack[i] = 0;
    __threadfence();
    __syncthreads();

    int sp = BF_ - 1;  // live only in threads 0..127 (one per batch element)

    int wave = tid >> 5, lane = tid & 31;
    int half = lane >> 4, r = lane & 15;
    int mtile = wave >> 2;
    int n0    = (wave & 3) * 4;
    int abatch = mtile * 16 + r;    // batch row this lane supplies to A fragments

    for (int t = 0; t < T_DIM; ++t) {
        if (tid == 0) { s_any[0] = 0; s_any[1] = 0; }
        __syncthreads();

        // --- stack automaton: one thread per batch element ---
        if (tid < B_DIM) {
            int b = tid;
            int a = arities[t * B_DIM + b];
            int* stk = stack + b * (T_DIM + BF_);
#pragma unroll
            for (int i = 0; i < BF_; ++i) {
                int valid = (a > i);
                s_idx[i][b]   = valid ? stk[sp - i] : 0;
                s_valid[i][b] = valid;
                if (valid) atomicOr(&s_any[i], 1);
            }
            int aabs = (a < 0) ? -a : a;
            sp = sp - aabs + 1;
            if (a != -1) stk[sp] = t;
        }
        __syncthreads();

        // --- children contributions: Delta = sum_i e_i @ Wc_i ---
        float* Mt = M + (size_t)t * B_DIM * DOUT_;
        v8f acc[4] = {};
#pragma unroll
        for (int i = 0; i < BF_; ++i) {
            if (!s_any[i]) continue;  // uniform: leaf steps skip the matmul
            const float* erow =
                s_valid[i][abatch]
                    ? (M + (size_t)s_idx[i][abatch] * B_DIM * DOUT_ + abatch * DOUT_)
                    : (const float*)nullptr;   // masked rows contribute zeros
            const __bf16* w = sWc + i * DOUT_ * DOUT_;
            for (int kc = 0; kc < 8; ++kc) {
                v16bf a = load_a_frag_f32(erow, kc, half);
#pragma unroll
                for (int j = 0; j < 4; ++j) {
                    v16bf b = load_b_frag_lds(w, n0 + j, kc, half, r);
                    acc[j] = __builtin_amdgcn_wmma_f32_16x16x32_bf16(
                        false, a, false, b, (short)0, acc[j], false, false);
                }
            }
        }

        // --- mem[t] = tanh(P[t] + Delta), in place ---
#pragma unroll
        for (int j = 0; j < 4; ++j) {
            int col = (n0 + j) * 16 + r;
#pragma unroll
            for (int v = 0; v < 8; ++v) {
                int row = mtile * 16 + half * 8 + v;
                float h = Mt[row * DOUT_ + col] + acc[j][v];
                Mt[row * DOUT_ + col] = tanhf(h);
            }
        }
        __threadfence();
        __syncthreads();
    }

    // --- emit memory[stack[b][sp]][b][:] ---
    if (tid < B_DIM) s_root[tid] = stack[tid * (T_DIM + BF_) + sp];
    __syncthreads();
    for (int e = tid; e < B_DIM * DOUT_; e += 1024) {
        int b = e >> 8, d = e & 255;
        out[e] = M[(size_t)s_root[b] * B_DIM * DOUT_ + b * DOUT_ + d];
    }
}

// ---- launch ----------------------------------------------------------------

extern "C" void kernel_launch(void* const* d_in, const int* in_sizes, int n_in,
                              void* d_out, int out_size, void* d_ws, size_t ws_size,
                              hipStream_t stream) {
    const float* x    = (const float*)d_in[0];   // [T,B,DIN] f32
    const int*   ar   = (const int*)  d_in[1];   // [T,B] i32
    const float* Wx   = (const float*)d_in[2];   // [DIN,DOUT] f32
    const float* Wc   = (const float*)d_in[3];   // [BF,DOUT,DOUT] f32
    const float* bias = (const float*)d_in[4];   // [DOUT] f32
    float* out = (float*)d_out;                  // [B,DOUT] f32

    char* ws = (char*)d_ws;
    size_t offM = (size_t)T_DIM * B_DIM * DOUT_ * sizeof(float);     // 128 MB
    float*          M   = (float*)ws;
    __hip_bfloat16* WxT = (__hip_bfloat16*)(ws + offM);
    __hip_bfloat16* WcT = (__hip_bfloat16*)(ws + offM + (size_t)DIN_ * DOUT_ * 2);
    int*            stk = (int*)(ws + offM + (size_t)DIN_ * DOUT_ * 2
                                          + (size_t)BF_ * DOUT_ * DOUT_ * 2);

    const size_t scan_lds = (size_t)BF_ * DOUT_ * DOUT_ * 2;         // 256 KB dynamic LDS

    convert_weights<<<(BF_ * DOUT_ * DOUT_ + 255) / 256, 256, 0, stream>>>(Wx, Wc, WxT, WcT);
    xwx_kernel<<<T_DIM, 1024, 0, stream>>>(x, WxT, bias, M);
    scan_kernel<<<1, 1024, scan_lds, stream>>>(ar, WcT, M, stk, out);
}
